// SparseCostRegNet_31190052503634
// MI455X (gfx1250) — matrix operand order repaired and checked by reading, not compiled, not run on hardware
//
#include <hip/hip_runtime.h>

// ---------------------------------------------------------------------------
// SparseCostRegNet (dense 160^3 3D U-Net) on MI455X / gfx1250.
//
// Memory-bound (~39 GFLOP, all activations fit in the 192 MB L2) -> f16
// activations, implicit GEMM on v_wmma_f32_16x16x32_f16, f32 accumulation,
// BN folded into weights, ReLU(+skip) fused into the epilogue.
//
// Hot-path structure:
//  * predication by ADDRESS via pure AND-masks (no bools, no cndmask chains):
//    off = (soff + pxb[tx] + cibb) & xm[tx] & zym ; invalid lanes get offset 0
//    which points at a zeroed guard block at ws+0,
//  * single scalar base + bounded 32-bit byte offsets (mask 0x1FFFFFF0 gives
//    the backend known-bits for addressing-mode folding),
//  * depth-1 software pipeline over the fully unrolled K-chunks,
//  * wave-uniform tile coords forced scalar via readfirstlane; DIN templated.
// ---------------------------------------------------------------------------

typedef _Float16 half_t;
typedef __attribute__((ext_vector_type(16))) _Float16 v16h;
typedef __attribute__((ext_vector_type(8)))  _Float16 v8h;
typedef __attribute__((ext_vector_type(8)))  float    v8f;

// mask of all representable workspace byte offsets (<512MB, 16B aligned)
#define WS_OFF_MASK 0x1FFFFFF0u

// ---------------------------------------------------------------------------
// ws guard block (invalid lanes read zeros from here) + input f32->f16
// ---------------------------------------------------------------------------
__global__ void zero_guard_kernel(unsigned* __restrict__ p) {
  p[threadIdx.x] = 0u;  // 64 threads * 4B = 256B
}

__global__ void f32_to_f16_kernel(const float* __restrict__ src,
                                  half_t* __restrict__ dst, int n) {
  int i = blockIdx.x * blockDim.x + threadIdx.x;
  if (i < n) dst[i] = (half_t)src[i];
}

// ---------------------------------------------------------------------------
// Weight repack: (3,3,3,Cin,Cout) f32 -> BN-folded f16 B fragments.
// Layout: [ntile][chunk][lane][e], lane = khalf*16 + n,
// element e -> K = chunk*32 + (e&8)*2 + khalf*8 + (e&7) ; tap = K>>lg(Cin), ci = K&(Cin-1).
// ---------------------------------------------------------------------------
template<int CIN, int COUT>
__global__ void pack_weights_kernel(const float* __restrict__ W,
                                    const float* __restrict__ g,
                                    const float* __restrict__ b,
                                    const float* __restrict__ m,
                                    const float* __restrict__ v,
                                    half_t* __restrict__ wfrag,
                                    float* __restrict__ bias) {
  constexpr int NTILES = (COUT + 15) / 16;
  constexpr int KMAX   = 27 * CIN;
  constexpr int NCHUNK = (KMAX + 31) / 32;
  constexpr int LG     = (CIN == 8) ? 3 : (CIN == 16) ? 4 : (CIN == 32) ? 5 : 6;
  const int idx = blockIdx.x * blockDim.x + threadIdx.x;
  if (idx < COUT) {
    float s = g[idx] * rsqrtf(v[idx] + 1e-5f);
    bias[idx] = b[idx] - m[idx] * s;
  }
  const int total = NCHUNK * NTILES * 32 * 16;
  if (idx >= total) return;
  const int e     = idx & 15;
  const int lane  = (idx >> 4) & 31;
  const int rest  = idx >> 9;           // = nt*NCHUNK + c
  const int c     = rest % NCHUNK;
  const int nt    = rest / NCHUNK;
  const int khalf = lane >> 4;
  const int n     = lane & 15;
  const int ng    = nt * 16 + n;
  const int K     = c * 32 + (e & 8) * 2 + khalf * 8 + (e & 7);
  float val = 0.0f;
  if (K < KMAX && ng < COUT) {
    const int tap = K >> LG;
    const int ci  = K & (CIN - 1);
    const float s = g[ng] * rsqrtf(v[ng] + 1e-5f);
    val = W[((size_t)tap * CIN + ci) * COUT + ng] * s;
  }
  wfrag[idx] = (half_t)val;
}

// ---------------------------------------------------------------------------
// Per-lane context precomputed once per wave (hoisted out of the K loop).
// pxb[t]: byte offset contribution of the x coordinate for tap-x = t.
// xm[t] : x validity as an AND-mask (0 or ~0).
// cibb  : khalf channel-run byte offset (0 for CIN==8 where ci is always 0).
// kmE/kmO: K-tail masks (khalf==0 / khalf==1 lanes live).
// ---------------------------------------------------------------------------
struct LaneCtx {
  unsigned pxb[3];
  unsigned xm[3];
  unsigned cibb;
  unsigned kmE, kmO;
  bool     kh;        // khalf != 0
};

// scalar (wave-uniform) z/y contribution + validity mask for one tap
template<bool DECONV, int STRIDE, int DIN, int CIN, int TZ, int TY, int CI>
__device__ __forceinline__ void zy_off(int oz, int oy, unsigned in_off,
                                       unsigned& soff, unsigned& zym) {
  int iz, iy; bool ok;
  if constexpr (DECONV) {
    const int uz = oz + 1 - TZ, uy = oy + 1 - TY;
    ok = (((uz | uy) & 1) == 0);               // parity (also kills -1)
    iz = uz >> 1; iy = uy >> 1;
    ok = ok && ((unsigned)iz < (unsigned)DIN) && ((unsigned)iy < (unsigned)DIN);
  } else if constexpr (STRIDE == 1) {
    iz = oz + (TZ - 1); iy = oy + (TY - 1);    // SAME pad 1
    ok = ((unsigned)iz < (unsigned)DIN) && ((unsigned)iy < (unsigned)DIN);
  } else {
    iz = 2 * oz + TZ; iy = 2 * oy + TY;        // SAME s2: pad_lo 0
    ok = (iz < DIN) && (iy < DIN);
  }
  soff = in_off + (unsigned)(((iz * DIN) + iy) * DIN * CIN + CI) * 2u;
  zym  = ok ? WS_OFF_MASK : 0u;                // doubles as known-bits hint
}

// ---------------------------------------------------------------------------
// One 8-channel (16B) K-run of the A fragment; KE = compile-time K of the
// khalf==0 lanes (khalf==1 lanes use KE+8). Invalid lanes get offset 0, which
// points at the zeroed guard block -> they load zeros; no data masking.
// ---------------------------------------------------------------------------
template<int CIN, bool DECONV, int STRIDE, int DIN, int KE>
__device__ __forceinline__ v8h load_a(const half_t* __restrict__ wsb,
                                      unsigned in_off, int oz, int oy,
                                      const LaneCtx& lc) {
  constexpr int LG   = (CIN == 8) ? 3 : (CIN == 16) ? 4 : (CIN == 32) ? 5 : 6;
  constexpr int KMAX = 27 * CIN;
  constexpr int KO   = KE + 8;
  constexpr bool VE  = KE < KMAX;
  constexpr bool VO  = KO < KMAX;
  if constexpr (!VE && !VO) {
    v8h z = {};
    return z;
  } else {
    constexpr int TE = (VE ? KE : KO) >> LG;
    constexpr int TO = (VO ? KO : KE) >> LG;
    constexpr int tzE = TE / 9, tyE = (TE % 9) / 3, txE = TE % 3;
    constexpr int tzO = TO / 9, tyO = (TO % 9) / 3, txO = TO % 3;
    constexpr int ciE = KE & (CIN - 1);   // khalf contribution lives in cibb
    constexpr bool SAME = (TE == TO);

    unsigned off;
    unsigned soffE, zymE;
    zy_off<DECONV, STRIDE, DIN, CIN, tzE, tyE, ciE>(oz, oy, in_off, soffE, zymE);
    if constexpr (SAME) {
      off = (soffE + lc.pxb[txE] + lc.cibb) & lc.xm[txE] & zymE;
    } else {
      unsigned soffO, zymO;
      zy_off<DECONV, STRIDE, DIN, CIN, tzO, tyO, ciE>(oz, oy, in_off, soffO, zymO);
      const unsigned offE = (soffE + lc.pxb[txE] + lc.cibb) & lc.xm[txE] & zymE;
      const unsigned offO = (soffO + lc.pxb[txO] + lc.cibb) & lc.xm[txO] & zymO;
      off = lc.kh ? offO : offE;
    }
    if constexpr (VE && !VO) off &= lc.kmE;    // K tail: only khalf==0 live
    if constexpr (!VE && VO) off &= lc.kmO;

    return *(const v8h*)((const char*)wsb + off);  // 0 -> zero guard block
  }
}

// ---------------------------------------------------------------------------
// Depth-1 software-pipelined, fully unrolled K-chunk loop.
// ---------------------------------------------------------------------------
struct Frag { v8h a0, a1; v16h b; };

template<int CIN, bool DECONV, int STRIDE, int DIN, int C>
__device__ __forceinline__ Frag load_frag(const half_t* __restrict__ wsb,
                                          unsigned in_off,
                                          const half_t* __restrict__ wfnt,
                                          int oz, int oy, const LaneCtx& lc,
                                          unsigned laneB) {
  Frag f;
  f.a0 = load_a<CIN, DECONV, STRIDE, DIN, 32 * C>(wsb, in_off, oz, oy, lc);
  f.a1 = load_a<CIN, DECONV, STRIDE, DIN, 32 * C + 16>(wsb, in_off, oz, oy, lc);
  f.b  = *(const v16h*)(wfnt + laneB + C * 512);   // saddr + imm offset
  return f;
}

__device__ __forceinline__ v8f do_wmma(const Frag& f, v8f acc) {
  union { v16h v; v8h h[2]; } A;
  A.h[0] = f.a0; A.h[1] = f.a1;
  return __builtin_amdgcn_wmma_f32_16x16x32_f16(
      /*neg_a=*/false, A.v, /*neg_b=*/false, f.b,
      /*c_mod=*/(short)0, acc, /*reuse_a=*/false, /*reuse_b=*/false);
}

template<int CIN, bool DECONV, int STRIDE, int DIN, int NCHUNK, int C>
__device__ __forceinline__ void chunk_rec(v8f& acc, Frag cur,
                                          const half_t* __restrict__ wsb,
                                          unsigned in_off,
                                          const half_t* __restrict__ wfnt,
                                          int oz, int oy, const LaneCtx& lc,
                                          unsigned laneB) {
  if constexpr (C + 1 < NCHUNK) {
    // issue next chunk's loads before consuming the current one
    Frag nxt = load_frag<CIN, DECONV, STRIDE, DIN, C + 1>(wsb, in_off, wfnt,
                                                          oz, oy, lc, laneB);
    acc = do_wmma(cur, acc);
    chunk_rec<CIN, DECONV, STRIDE, DIN, NCHUNK, C + 1>(acc, nxt, wsb, in_off,
                                                       wfnt, oz, oy, lc, laneB);
  } else {
    acc = do_wmma(cur, acc);
  }
}

// ---------------------------------------------------------------------------
// Implicit-GEMM conv3d / conv_transpose3d, fused BN+ReLU(+skip) epilogue.
// blockDim = (32, 8): 8 waves per workgroup, one 16x16 output tile per wave.
// ---------------------------------------------------------------------------
template<int CIN, int COUT, bool DECONV, int STRIDE, int DIN,
         bool HAS_SKIP, bool OUT_F32>
__global__ __launch_bounds__(256, 2)
void conv3d_wmma_kernel(const half_t* __restrict__ wsbase, unsigned in_off,
                        const half_t* __restrict__ wfrag,
                        const float*  __restrict__ bias,
                        const half_t* __restrict__ skip,
                        half_t* __restrict__ outh,
                        float*  __restrict__ outf) {
  constexpr int DOUT   = DECONV ? DIN * 2 : (STRIDE == 2 ? DIN / 2 : DIN);
  constexpr int NTILES = (COUT + 15) / 16;
  constexpr int NCHUNK = (27 * CIN + 31) / 32;
  constexpr int XTILES = (DOUT + 15) / 16;
  constexpr bool XFULL = (DOUT % 16) == 0;

  const int lane = threadIdx.x;  // 0..31
  // wave-uniform tile id forced scalar: oz/oy/xt/nt + all z/y math in SGPRs
  const int wid = blockIdx.x * 8 + __builtin_amdgcn_readfirstlane(threadIdx.y);
  constexpr int TOTAL = DOUT * DOUT * XTILES * NTILES;
  if (wid >= TOTAL) return;  // wave-uniform -> EXEC stays full for WMMA

  int nt = wid % NTILES;
  int t  = wid / NTILES;
  const int xt = t % XTILES;  t /= XTILES;
  const int oy = t % DOUT;
  const int oz = t / DOUT;

  const int  khalf = lane >> 4;
  const int  mrow  = lane & 15;            // A-fragment row (output voxel)
  const int  ox    = xt * 16 + mrow;
  const bool oxv   = XFULL || (ox < DOUT);

  // per-lane x offsets/validity masks for the three tap-x positions (hoisted)
  LaneCtx lc;
  lc.kh   = (khalf != 0);
  lc.cibb = (CIN == 8) ? 0u : (unsigned)(khalf * 16);
  lc.kmE  = khalf ? 0u : ~0u;
  lc.kmO  = khalf ? ~0u : 0u;
#pragma unroll
  for (int tx = 0; tx < 3; ++tx) {
    int ix; bool ok = oxv;
    if (DECONV) {
      const int ux = ox + 1 - tx;
      ok = ok && ((ux & 1) == 0);
      ix = ux >> 1;
      ok = ok && ((unsigned)ix < (unsigned)DIN);
    } else if (STRIDE == 1) {
      ix = ox + (tx - 1);
      ok = ok && ((unsigned)ix < (unsigned)DIN);
    } else {
      ix = 2 * ox + tx;
      ok = ok && (ix < DIN);
    }
    lc.pxb[tx] = (unsigned)(ix * CIN * 2);
    lc.xm[tx]  = ok ? ~0u : 0u;
  }

  const half_t* wfnt  = wfrag + (size_t)nt * NCHUNK * 512;  // [nt][c][lane][e]
  const unsigned laneB = (unsigned)lane * 16;

  v8f acc = {};
  Frag f0 = load_frag<CIN, DECONV, STRIDE, DIN, 0>(wsbase, in_off, wfnt,
                                                   oz, oy, lc, laneB);
  chunk_rec<CIN, DECONV, STRIDE, DIN, NCHUNK, 0>(acc, f0, wsbase, in_off, wfnt,
                                                 oz, oy, lc, laneB);

  // C/D layout: VGPR r -> M = r + 8*khalf, N = lane&15.
  const int n  = lane & 15;
  const int ng = nt * 16 + n;
  if ((COUT % 16 == 0) || (ng < COUT)) {
    const float bv = bias[ng];
#pragma unroll
    for (int r = 0; r < 8; ++r) {
      const int oxs = xt * 16 + khalf * 8 + r;
      if (XFULL || (oxs < DOUT)) {
        const size_t vox = (((size_t)oz * DOUT) + oy) * DOUT + oxs;
        float val = fmaxf(acc[r] + bv, 0.0f);
        if (HAS_SKIP) val += (float)skip[vox * COUT + ng];
        if (OUT_F32) {
          // write-once 131MB f32 result: keep it out of the L2 working set
          __builtin_nontemporal_store(val, &outf[vox * COUT + ng]);
        } else {
          outh[vox * COUT + ng] = (half_t)val;
        }
      }
    }
  }
}

// ---------------------------------------------------------------------------
// Host-side launch helpers
// ---------------------------------------------------------------------------
template<int CIN, int COUT>
static void run_pack(const float* W, const float* g, const float* b,
                     const float* m, const float* v, half_t* wf, float* bias,
                     hipStream_t s) {
  const int elems = ((27 * CIN + 31) / 32) * ((COUT + 15) / 16) * 512;
  pack_weights_kernel<CIN, COUT><<<(elems + 255) / 256, 256, 0, s>>>(
      W, g, b, m, v, wf, bias);
}

template<int CIN, int COUT, bool DECONV, int STRIDE, int DIN,
         bool HAS_SKIP, bool OUT_F32>
static void run_layer(const half_t* wsbase, const half_t* in, const half_t* wf,
                      const float* bias, const half_t* skip, half_t* outh,
                      float* outf, hipStream_t s) {
  constexpr int DOUT = DECONV ? DIN * 2 : (STRIDE == 2 ? DIN / 2 : DIN);
  const long long waves =
      (long long)DOUT * DOUT * ((DOUT + 15) / 16) * ((COUT + 15) / 16);
  const unsigned in_off = (unsigned)((const char*)in - (const char*)wsbase);
  dim3 blk(32, 8);
  const unsigned nb = (unsigned)((waves + 7) / 8);
  conv3d_wmma_kernel<CIN, COUT, DECONV, STRIDE, DIN, HAS_SKIP, OUT_F32>
      <<<nb, blk, 0, s>>>(wsbase, in_off, wf, bias, skip, outh, outf);
}

static size_t frag_elems(int cin, int cout) {
  return (size_t)((27 * cin + 31) / 32) * ((cout + 15) / 16) * 512;
}

extern "C" void kernel_launch(void* const* d_in, const int* in_sizes, int n_in,
                              void* d_out, int out_size, void* d_ws, size_t ws_size,
                              hipStream_t stream) {
  (void)in_sizes; (void)n_in; (void)out_size; (void)ws_size;
  const float* x = (const float*)d_in[0];
  auto P = [&](int layer, int j) { return (const float*)d_in[1 + layer * 5 + j]; };

  // ---- workspace layout (bytes, 256B aligned; first 256B = zero guard) ----
  char* ws = (char*)d_ws;
  half_t* wsb = (half_t*)ws;
  size_t off = 256;                               // guard block at offset 0
  auto alloc = [&](size_t bytes) {
    size_t o = off; off += (bytes + 255) & ~(size_t)255; return o;
  };
  const size_t V160 = 160ull * 160 * 160, V80 = 80ull * 80 * 80;
  const size_t V40  = 40ull  * 40  * 40,  V20 = 20ull * 20 * 20;

  half_t* X0 = (half_t*)(ws + alloc(V160 * 8  * sizeof(half_t)));
  half_t* c0 = (half_t*)(ws + alloc(V160 * 8  * sizeof(half_t)));
  half_t* tA = (half_t*)(ws + alloc(V80  * 16 * sizeof(half_t)));  // t1, later d9
  half_t* c2 = (half_t*)(ws + alloc(V80  * 16 * sizeof(half_t)));
  half_t* tB = (half_t*)(ws + alloc(V40  * 32 * sizeof(half_t)));  // t3, later d7
  half_t* c4 = (half_t*)(ws + alloc(V40  * 32 * sizeof(half_t)));
  half_t* t5 = (half_t*)(ws + alloc(V20  * 64 * sizeof(half_t)));
  half_t* t6 = (half_t*)(ws + alloc(V20  * 64 * sizeof(half_t)));

  static const int ci_[10] = {8, 8, 16, 16, 32, 32, 64, 64, 32, 16};
  static const int co_[10] = {8, 16, 16, 32, 32, 64, 64, 32, 16, 8};
  half_t* wf[10]; float* bs[10];
  for (int l = 0; l < 10; ++l)
    wf[l] = (half_t*)(ws + alloc(frag_elems(ci_[l], co_[l]) * sizeof(half_t)));
  for (int l = 0; l < 10; ++l)
    bs[l] = (float*)(ws + alloc(64 * sizeof(float)));

  // ---- zero the guard block (invalid lanes read zeros from ws[0..255]) ----
  zero_guard_kernel<<<1, 64, 0, stream>>>((unsigned*)ws);

  // ---- fold BN into weights (device-side repack) ----
  run_pack<8,  8 >(P(0,0), P(0,1), P(0,2), P(0,3), P(0,4), wf[0], bs[0], stream);
  run_pack<8,  16>(P(1,0), P(1,1), P(1,2), P(1,3), P(1,4), wf[1], bs[1], stream);
  run_pack<16, 16>(P(2,0), P(2,1), P(2,2), P(2,3), P(2,4), wf[2], bs[2], stream);
  run_pack<16, 32>(P(3,0), P(3,1), P(3,2), P(3,3), P(3,4), wf[3], bs[3], stream);
  run_pack<32, 32>(P(4,0), P(4,1), P(4,2), P(4,3), P(4,4), wf[4], bs[4], stream);
  run_pack<32, 64>(P(5,0), P(5,1), P(5,2), P(5,3), P(5,4), wf[5], bs[5], stream);
  run_pack<64, 64>(P(6,0), P(6,1), P(6,2), P(6,3), P(6,4), wf[6], bs[6], stream);
  run_pack<64, 32>(P(7,0), P(7,1), P(7,2), P(7,3), P(7,4), wf[7], bs[7], stream);
  run_pack<32, 16>(P(8,0), P(8,1), P(8,2), P(8,3), P(8,4), wf[8], bs[8], stream);
  run_pack<16, 8 >(P(9,0), P(9,1), P(9,2), P(9,3), P(9,4), wf[9], bs[9], stream);

  // ---- input f32 -> f16 channels-last ----
  {
    const int n = (int)(V160 * 8);
    f32_to_f16_kernel<<<(n + 255) / 256, 256, 0, stream>>>(x, X0, n);
  }

  // ---- encoder ----
  run_layer<8,  8,  false, 1, 160, false, false>(wsb, X0, wf[0], bs[0], nullptr, c0, nullptr, stream);
  run_layer<8,  16, false, 2, 160, false, false>(wsb, c0, wf[1], bs[1], nullptr, tA, nullptr, stream);
  run_layer<16, 16, false, 1, 80,  false, false>(wsb, tA, wf[2], bs[2], nullptr, c2, nullptr, stream);
  run_layer<16, 32, false, 2, 80,  false, false>(wsb, c2, wf[3], bs[3], nullptr, tB, nullptr, stream);
  run_layer<32, 32, false, 1, 40,  false, false>(wsb, tB, wf[4], bs[4], nullptr, c4, nullptr, stream);
  run_layer<32, 64, false, 2, 40,  false, false>(wsb, c4, wf[5], bs[5], nullptr, t5, nullptr, stream);
  run_layer<64, 64, false, 1, 20,  false, false>(wsb, t5, wf[6], bs[6], nullptr, t6, nullptr, stream);

  // ---- decoder (transposed convs with fused skip adds) ----
  half_t* d7 = tB;   // t3 region is dead after conv4
  half_t* d9 = tA;   // t1 region is dead after conv2
  run_layer<64, 32, true, 2, 20, true, false>(wsb, t6, wf[7], bs[7], c4, d7, nullptr, stream);
  run_layer<32, 16, true, 2, 40, true, false>(wsb, d7, wf[8], bs[8], c2, d9, nullptr, stream);
  run_layer<16, 8,  true, 2, 80, true, true >(wsb, d9, wf[9], bs[9], c0, nullptr,
                                              (float*)d_out, stream);
}
